// NCAModel_21397527068823
// MI455X (gfx1250) — matrix operand order repaired and checked by reading, not compiled, not run on hardware
//
#include <hip/hip_runtime.h>
#include <math.h>

typedef __attribute__((ext_vector_type(16))) _Float16 v16h;
typedef __attribute__((ext_vector_type(8)))  _Float16 v8h;
typedef __attribute__((ext_vector_type(8)))  float    v8f;

#define CH   16
#define HH   256
#define WW   256
#define BB   32
#define TPX  64     // pixels per workgroup (one strip of a row)
#define NPT  4      // 16-pixel tiles per workgroup

// ---------------------------------------------------------------------------
// Kernel 0: convert W1/W2 (f32) into f16 WMMA A-operand layouts in d_ws.
// w1A: [og(8)][chunk(2)][lane(32)][hi(16)]   (K padded 48 -> 64)
// w2A: [chunk(4)][lane(32)][hi(16)]
// A layout packed here (16x32 f16, contiguous k, B uses the same): lanes 0-15: M=lane, halves 0-15 -> K 0-15
//                       lanes 16-31: M=lane-16, halves 0-15 -> K 16-31
// ---------------------------------------------------------------------------
__launch_bounds__(256)
__global__ void nca_prep_weights(const float* __restrict__ w1,
                                 const float* __restrict__ w2,
                                 _Float16* __restrict__ w1A,
                                 _Float16* __restrict__ w2A) {
  int tid = threadIdx.x;
  for (int i = tid; i < 8 * 2 * 32 * 16; i += 256) {
    int hi    = i & 15;
    int lane  = (i >> 4) & 31;
    int chunk = (i >> 9) & 1;
    int og    = i >> 10;
    int g = lane >> 4, n = lane & 15;
    int k = chunk * 32 + g * 16 + hi;
    float v = (k < 48) ? w1[(og * 16 + n) * 48 + k] : 0.0f;
    w1A[i] = (_Float16)v;
  }
  for (int i = tid; i < 4 * 32 * 16; i += 256) {
    int hi   = i & 15;
    int lane = (i >> 4) & 31;
    int cc   = i >> 9;
    int g = lane >> 4, n = lane & 15;
    int k = cc * 32 + g * 16 + hi;
    w2A[i] = (_Float16)w2[n * 128 + k];
  }
}

// ---------------------------------------------------------------------------
// Kernel 1: fused perception (depthwise 3x3, wrap) + MLP (48->128 relu ->16)
//           + stochastic update. Writes x_new to out, new alpha plane to ws.
// ---------------------------------------------------------------------------
__launch_bounds__(256)
__global__ void nca_update_kernel(const float* __restrict__ x,
                                  const float* __restrict__ rand01,
                                  const _Float16* __restrict__ w1A,
                                  const float* __restrict__ b1,
                                  const _Float16* __restrict__ w2A,
                                  const float* __restrict__ b2,
                                  float* __restrict__ xnew,
                                  float* __restrict__ alpha_ws) {
  // x rows r-1..r+1 for 66 columns (64 + wrap halo), f32
  __shared__ __align__(16) float xs[3][CH][TPX + 8];
  // perception features in WMMA B-operand layout: [tile][Kchunk32][lane][16 halves]
  __shared__ __align__(32) _Float16 yB[NPT][2][32][16];
  // hidden layer in WMMA B-operand layout: K=128 -> 4 chunks
  __shared__ __align__(32) _Float16 hB[NPT][4][32][16];

  const int tid = threadIdx.x;
  const int blk = blockIdx.x;
  const int xt  = blk & 3;
  const int r   = (blk >> 2) & (HH - 1);
  const int b   = blk >> 10;
  const int x0  = xt * TPX;

  // warm L2/L0 for the rand row consumed in the epilogue (global_prefetch_b8)
  __builtin_prefetch(&rand01[(b * HH + r) * WW + x0 + (tid & 63)], 0, 3);

  // ---- phase 0a: zero yB (covers the K=48..63 pad region) ----
  {
    float* yz = (float*)&yB[0][0][0][0];          // 8192 B = 2048 dwords
    for (int i = tid; i < 2048; i += 256) yz[i] = 0.0f;
  }
  // ---- phase 0b: stage x rows (wrap in H and W) ----
  for (int i = tid; i < 3 * CH * 66; i += 256) {
    int j   = i % 66;
    int c   = (i / 66) % CH;
    int rr3 = i / (66 * CH);
    int gr  = (r + rr3 - 1 + HH) & (HH - 1);
    int gp  = (x0 + j - 1 + WW) & (WW - 1);
    xs[rr3][c][j] = x[(((b * CH) + c) * HH + gr) * WW + gp];
  }
  __syncthreads();

  // ---- phase 1: perception -> yB in f16 B-layout ----
  #pragma unroll
  for (int it = 0; it < 4; ++it) {
    int task = it * 256 + tid;          // 1024 = 64 px * 16 ch
    int p = task & 63;
    int c = task >> 6;
    int j = p + 1;
    float m00 = xs[0][c][j - 1], m01 = xs[0][c][j], m02 = xs[0][c][j + 1];
    float m10 = xs[1][c][j - 1], m11 = xs[1][c][j], m12 = xs[1][c][j + 1];
    float m20 = xs[2][c][j - 1], m21 = xs[2][c][j], m22 = xs[2][c][j + 1];
    float vid = m11;
    float vdx = ((m02 - m00) + 2.0f * (m12 - m10) + (m22 - m20)) * 0.125f;
    float vdy = ((m20 - m00) + 2.0f * (m21 - m01) + (m22 - m02)) * 0.125f;
    int pt = p >> 4, n = p & 15;
    float vals[3] = {vid, vdx, vdy};
    #pragma unroll
    for (int q = 0; q < 3; ++q) {
      int K = c * 3 + q;
      int chunk = K >> 5, kk = K & 31;
      yB[pt][chunk][n + ((kk >> 4) << 4)][kk & 15] = (_Float16)vals[q];
    }
  }
  __syncthreads();

  const int wave = tid >> 5;
  const int lane = tid & 31;
  const int g    = lane >> 4;     // half-wave: K sub-rows (A/B) / M half (D)
  const int n    = lane & 15;     // A-row / B-column / D-column index
  const int og   = wave;          // layer-1 output-channel group (16 channels)

  // ---- phase 2: layer 1 GEMM; A tiles preconverted in d_ws (b128 loads) ----
  v16h A0 = *(const v16h*)(w1A + (((og * 2 + 0) * 32) + lane) * 16);
  v16h A1 = *(const v16h*)(w1A + (((og * 2 + 1) * 32) + lane) * 16);
  v8f bias1;
  #pragma unroll
  for (int rr = 0; rr < 8; ++rr) bias1[rr] = b1[og * 16 + g * 8 + rr];

  #pragma unroll
  for (int pt = 0; pt < NPT; ++pt) {
    v16h B0 = *(const v16h*)&yB[pt][0][lane][0];
    v16h B1 = *(const v16h*)&yB[pt][1][lane][0];
    v8f acc = bias1;
    acc = __builtin_amdgcn_wmma_f32_16x16x32_f16(false, A0, false, B0,
                                                 (short)0, acc, false, false);
    acc = __builtin_amdgcn_wmma_f32_16x16x32_f16(false, A1, false, B1,
                                                 (short)0, acc, false, false);
    // ReLU + convert; scatter into layer-2 B layout (8 halves = 16B store)
    v8h hv;
    #pragma unroll
    for (int rr = 0; rr < 8; ++rr) hv[rr] = (_Float16)fmaxf(acc[rr], 0.0f);
    *(v8h*)&hB[pt][og >> 1][n + ((og & 1) << 4)][g * 8] = hv;
  }
  __syncthreads();

  // ---- phase 3: layer 2 GEMM (16 out ch, K=128) + stochastic update ----
  if (wave < NPT) {
    const int pt = wave;
    v16h A2[4];
    #pragma unroll
    for (int cc = 0; cc < 4; ++cc)
      A2[cc] = *(const v16h*)(w2A + ((cc * 32) + lane) * 16);

    v8f acc;
    #pragma unroll
    for (int rr = 0; rr < 8; ++rr) acc[rr] = b2[g * 8 + rr];
    #pragma unroll
    for (int cc = 0; cc < 4; ++cc) {
      v16h Bv = *(const v16h*)&hB[pt][cc][lane][0];
      acc = __builtin_amdgcn_wmma_f32_16x16x32_f16(false, A2[cc], false, Bv,
                                                   (short)0, acc, false, false);
    }
    // epilogue: x_new = x + dx * (rand <= 0.5)
    int p  = pt * 16 + n;
    int px = x0 + p;
    float rv  = rand01[(b * HH + r) * WW + px];
    float upd = (rv <= 0.5f) ? 1.0f : 0.0f;
    #pragma unroll
    for (int rr = 0; rr < 8; ++rr) {
      int m   = g * 8 + rr;                 // channel (D-layout row)
      float xo = xs[1][m][p + 1];
      float xn = xo + acc[rr] * upd;
      xnew[((b * CH + m) * HH + r) * WW + px] = xn;
      if (m == 3) alpha_ws[(b * HH + r) * WW + px] = xn;
    }
  }
}

// ---------------------------------------------------------------------------
// Kernel 2: life mask. pre from original x alpha, post from ws alpha plane.
// 3x3 max-pool with -inf (edge-clamped) padding, then out *= mask in place.
// ---------------------------------------------------------------------------
__launch_bounds__(256)
__global__ void nca_mask_kernel(const float* __restrict__ x,
                                const float* __restrict__ alpha_ws,
                                float* __restrict__ out) {
  int blk = blockIdx.x;
  int r = blk & (HH - 1);
  int b = blk >> 8;
  int px = threadIdx.x;
  float preM = -INFINITY, postM = -INFINITY;
  #pragma unroll
  for (int dr = -1; dr <= 1; ++dr) {
    int rr = r + dr;
    if (rr < 0 || rr >= HH) continue;
    #pragma unroll
    for (int dp = -1; dp <= 1; ++dp) {
      int pp = px + dp;
      if (pp < 0 || pp >= WW) continue;
      preM  = fmaxf(preM,  x[((b * CH + 3) * HH + rr) * WW + pp]);
      postM = fmaxf(postM, alpha_ws[(b * HH + rr) * WW + pp]);
    }
  }
  float mask = (preM > 0.1f && postM > 0.1f) ? 1.0f : 0.0f;
  #pragma unroll
  for (int c = 0; c < CH; ++c) {
    size_t idx = (size_t)((b * CH + c) * HH + r) * WW + px;
    out[idx] *= mask;
  }
}

// ---------------------------------------------------------------------------
extern "C" void kernel_launch(void* const* d_in, const int* in_sizes, int n_in,
                              void* d_out, int out_size, void* d_ws, size_t ws_size,
                              hipStream_t stream) {
  const float* x   = (const float*)d_in[0];
  const float* rnd = (const float*)d_in[1];
  // d_in[2] = w_perc: fixed identity/Sobel kernels, hardcoded in kernel 1
  const float* w1  = (const float*)d_in[3];
  const float* b1  = (const float*)d_in[4];
  const float* w2  = (const float*)d_in[5];
  const float* b2  = (const float*)d_in[6];
  float* out = (float*)d_out;

  // d_ws layout: [ alpha plane: 32*256*256 f32 = 8 MB ][ w1A 16 KB ][ w2A 4 KB ]
  float*    aws = (float*)d_ws;
  _Float16* w1A = (_Float16*)((float*)d_ws + (size_t)BB * HH * WW);
  _Float16* w2A = w1A + 8 * 2 * 32 * 16;

  nca_prep_weights<<<1, 256, 0, stream>>>(w1, w2, w1A, w2A);

  dim3 grid1(BB * HH * (WW / TPX));   // 32768 workgroups
  nca_update_kernel<<<grid1, 256, 0, stream>>>(x, rnd, w1A, b1, w2A, b2, out, aws);

  dim3 grid2(BB * HH);                // 8192 workgroups
  nca_mask_kernel<<<grid2, 256, 0, stream>>>(x, aws, out);
}
